// TinyTraceRNN_13048110645866
// MI455X (gfx1250) — compile-verified, matrix operand order checked by reading, not run on hardware
//
#include <hip/hip_runtime.h>
#include <hip/hip_bf16.h>

typedef __attribute__((ext_vector_type(16))) _Float16 v16h;
typedef __attribute__((ext_vector_type(8)))  _Float16 v8h;
typedef __attribute__((ext_vector_type(8)))  float    v8f;

#define NB   128
#define NS   2048
#define NE   256
#define NH   256
// Padded LDS row strides (elements). 264 halves = 528B = 33*16B: rows stay
// 16B-aligned for b128 loads, and 132 DWORDs == 4 (mod 64 banks) keeps the
// 16-lane b128 fragment loads bank-conflict-free.
#define WSTR 264
#define SSTR 264
#define PSTR 264

union FragU { v16h v; v8h h[2]; };

// Hardware tanh (TRANS32 op on CDNA5) — confirmed present in round 2 (valu 1296->332).
__device__ __forceinline__ float fast_tanh(float x) {
#if __has_builtin(__builtin_amdgcn_tanhf)
  return __builtin_amdgcn_tanhf(x);
#elif __has_builtin(__builtin_amdgcn_tanh_f32)
  return __builtin_amdgcn_tanh_f32(x);
#else
  return tanhf(x);
#endif
}

// -------- setup: fp16 weight copies + P0 token table + fused layer-1 bias ----
__global__ void prep_kernel(const float* __restrict__ emb,
                            const float* __restrict__ Wih0,
                            const float* __restrict__ Whh0,
                            const float* __restrict__ bih0,
                            const float* __restrict__ bhh0,
                            const float* __restrict__ Wih1,
                            const float* __restrict__ Whh1,
                            const float* __restrict__ bih1,
                            const float* __restrict__ bhh1,
                            _Float16* __restrict__ w0g,
                            _Float16* __restrict__ w1g,
                            _Float16* __restrict__ w2g,
                            float* __restrict__ P0g,
                            float* __restrict__ b1g)
{
  int tid = blockIdx.x * blockDim.x + threadIdx.x;
  int nthreads = gridDim.x * blockDim.x;
  for (int i = tid; i < NH * NH; i += nthreads) {
    w0g[i] = (_Float16)Whh0[i];   // layer-0 recurrent
    w1g[i] = (_Float16)Wih1[i];   // layer-1 input
    w2g[i] = (_Float16)Whh1[i];   // layer-1 recurrent
  }
  if (blockIdx.x == 0 && threadIdx.x < NH) {
    int n = threadIdx.x;
    float b0v = bih0[n] + bhh0[n];
    for (int v = 0; v < 3; ++v) {
      float s = b0v;
      const float* er = emb + v * NE;
      const float* wr = Wih0 + n * NE;
      for (int e = 0; e < NE; ++e) s += er[e] * wr[e];
      P0g[v * NH + n] = s;        // P0[v][n] = (emb[v] . Wih0[n]) + biases
    }
    b1g[n] = bih1[n] + bhh1[n];
  }
}

// A fragment (16x32 f16, M x K): lane<16 -> K {k0..k0+7, k0+16..k0+23}, lane>=16 -> +8
__device__ __forceinline__ v16h load_a_frag(const _Float16* st, int lane, int c) {
  const _Float16* p = st + (lane & 15) * SSTR + 32 * c + 8 * (lane >> 4);
  FragU f;
  f.h[0] = *(const v8h*)(p);
  f.h[1] = *(const v8h*)(p + 16);
  return f.v;
}
// B fragment (32x16 f16, K x N), B[k][n] = W[n][k]: lane reads its W row (padded stride)
__device__ __forceinline__ v16h load_b_frag(const _Float16* w, int nb, int lane, int c) {
  const _Float16* p = w + (nb + (lane & 15)) * WSTR + 32 * c + 8 * (lane >> 4);
  FragU f;
  f.h[0] = *(const v8h*)(p);
  f.h[1] = *(const v8h*)(p + 16);
  return f.v;
}
// B fragment straight from (unpadded) global f16 weights, for VGPR-persistent slices
__device__ __forceinline__ v16h load_b_frag_g(const _Float16* w, int nb, int lane, int c) {
  const _Float16* p = w + (nb + (lane & 15)) * NH + 32 * c + 8 * (lane >> 4);
  FragU f;
  f.h[0] = *(const v8h*)(p);
  f.h[1] = *(const v8h*)(p + 16);
  return f.v;
}

// -------- persistent RNN cell: one WGP per 16 batch rows, full 2048-step loop ----
__global__ __launch_bounds__(512, 1)
void rnn_kernel(const int* __restrict__ tokens,
                const int* __restrict__ lengths,
                const _Float16* __restrict__ w0g,
                const _Float16* __restrict__ w1g,
                const _Float16* __restrict__ w2g,
                const float* __restrict__ P0g,
                const float* __restrict__ b1g,
                const float* __restrict__ Wc,
                const float* __restrict__ bc,
                float* __restrict__ out)
{
  __shared__ _Float16 w0lds[NH * WSTR];       // Whh0, 135168 B (shared B-frags)
  __shared__ _Float16 h0s[2][16 * SSTR];      // layer-0 state, double-buffered
  __shared__ _Float16 h1s[2][16 * SSTR];      // layer-1 state, double-buffered
  __shared__ float    P0lds[3 * PSTR];        // token table
  __shared__ int      tokLds[2][16];          // tokens for step t / t+1

  const int tid  = threadIdx.x;
  const int lane = tid & 31;
  const int wv   = tid >> 5;       // wave 0..15
  const int nb   = wv * 16;        // this wave's N-slice base
  const int nn   = lane & 15;
  const int half = lane >> 4;      // 0: M/K low half, 1: high half (per ISA layout)
  const int b0   = blockIdx.x * 16;

  // Stage Whh0 into LDS with padded rows (u32 copies keep half pairs intact).
  {
    const unsigned* s0 = (const unsigned*)w0g;
    unsigned* d0 = (unsigned*)w0lds;
    for (int i = tid; i < NH * (NH / 2); i += 512) {
      int n = i >> 7, kk = i & 127;
      d0[n * (WSTR / 2) + kk] = s0[i];
    }
  }
  for (int i = tid; i < 3 * NH; i += 512)
    P0lds[(i >> 8) * PSTR + (i & 255)] = P0g[i];
  {
    unsigned* z0 = (unsigned*)h0s[0];
    unsigned* z1 = (unsigned*)h1s[0];
    for (int i = tid; i < (16 * SSTR) / 2; i += 512) { z0[i] = 0u; z1[i] = 0u; }
  }
  if (tid < 16) tokLds[0][tid] = tokens[(b0 + tid) * NS];

  // Per-wave persistent B-fragments in VGPRs: Wih1 and Whh1 (each wave needs
  // only its own 16 N-columns -> 64 VGPRs per matrix, no per-step LDS traffic).
  v16h w1f[8], w2f[8];
#pragma unroll
  for (int c = 0; c < 8; ++c) {
    w1f[c] = load_b_frag_g(w1g, nb, lane, c);
    w2f[c] = load_b_frag_g(w2g, nb, lane, c);
  }

  const float b1v = b1g[nb + nn];

  int len8[8];
#pragma unroll
  for (int r = 0; r < 8; ++r) len8[r] = lengths[b0 + r + 8 * half];

  // Post-mask states held in registers (C/D element (r, lane) = (M=r+8*half, N=nb+nn))
  float keep0[8], keep1[8];
#pragma unroll
  for (int r = 0; r < 8; ++r) { keep0[r] = 0.f; keep1[r] = 0.f; }

  __syncthreads();

  int p = 0;
  for (int t = 0; t < NS; ++t) {
    const int q = p ^ 1;

    if (tid < 16) {                              // prefetch tokens for t+1 early
      int tn = (t + 1 < NS) ? (t + 1) : t;
      tokLds[q][tid] = tokens[(b0 + tid) * NS + tn];
    }

    // ---- layer 0: h0_new = tanh(P0[token] + h0_prev @ Whh0^T), masked ----
    // K-loop software-pipelined by one stage: LDS latency overlaps WMMA.
    v8f acc = {};
    {
      v16h a_c = load_a_frag(h0s[p], lane, 0);
      v16h b_c = load_b_frag(w0lds, nb, lane, 0);
#pragma unroll
      for (int c = 0; c < 8; ++c) {
        v16h a_n = a_c, b_n = b_c;
        if (c < 7) {
          a_n = load_a_frag(h0s[p], lane, c + 1);
          b_n = load_b_frag(w0lds, nb, lane, c + 1);
        }
        acc = __builtin_amdgcn_wmma_f32_16x16x32_f16(false, a_c, false, b_c,
                                                     (short)0, acc, false, false);
        a_c = a_n; b_c = b_n;
      }
    }
#pragma unroll
    for (int r = 0; r < 8; ++r) {
      int m = r + 8 * half;
      int tok = tokLds[p][m];
      float x0 = P0lds[tok * PSTR + nb + nn];
      float v = fast_tanh(acc[r] + x0);
      if (t < len8[r]) keep0[r] = v;            // freeze finished sequences
      h0s[q][m * SSTR + nb + nn] = (_Float16)keep0[r];
    }
    __syncthreads();

    // ---- layer 1: h1_new = tanh(b1 + h0_new @ Wih1^T + h1_prev @ Whh1^T) ----
    // Two independent accumulator chains (depth 8 each instead of 16),
    // interleaved, A-fragments pipelined one stage ahead.
    v8f accI, accH = {};
#pragma unroll
    for (int r = 0; r < 8; ++r) accI[r] = b1v;
    {
      v16h aI = load_a_frag(h0s[q], lane, 0);
      v16h aH = load_a_frag(h1s[p], lane, 0);
#pragma unroll
      for (int c = 0; c < 8; ++c) {
        v16h aIn = aI, aHn = aH;
        if (c < 7) {
          aIn = load_a_frag(h0s[q], lane, c + 1);
          aHn = load_a_frag(h1s[p], lane, c + 1);
        }
        accI = __builtin_amdgcn_wmma_f32_16x16x32_f16(false, aI, false, w1f[c],
                                                      (short)0, accI, false, false);
        accH = __builtin_amdgcn_wmma_f32_16x16x32_f16(false, aH, false, w2f[c],
                                                      (short)0, accH, false, false);
        aI = aIn; aH = aHn;
      }
    }
#pragma unroll
    for (int r = 0; r < 8; ++r) {
      int m = r + 8 * half;
      float v = fast_tanh(accI[r] + accH[r]);
      if (t < len8[r]) keep1[r] = v;
      h1s[q][m * SSTR + nb + nn] = (_Float16)keep1[r];
    }
    __syncthreads();
    p = q;
  }

  // ---- classifier head: logits[b] = h1[b] . Wc + bc ----
  if (tid < 16) {
    float s = bc[0];
    const _Float16* hr = &h1s[p][tid * SSTR];
    for (int n = 0; n < NH; ++n) s += (float)hr[n] * Wc[n];
    out[b0 + tid] = s;
  }
}

extern "C" void kernel_launch(void* const* d_in, const int* in_sizes, int n_in,
                              void* d_out, int out_size, void* d_ws, size_t ws_size,
                              hipStream_t stream) {
  (void)in_sizes; (void)n_in; (void)out_size; (void)ws_size;
  const int*   tokens  = (const int*)  d_in[0];
  const int*   lengths = (const int*)  d_in[1];
  const float* emb     = (const float*)d_in[2];
  const float* Wih0    = (const float*)d_in[3];
  const float* Whh0    = (const float*)d_in[4];
  const float* bih0    = (const float*)d_in[5];
  const float* bhh0    = (const float*)d_in[6];
  const float* Wih1    = (const float*)d_in[7];
  const float* Whh1    = (const float*)d_in[8];
  const float* bih1    = (const float*)d_in[9];
  const float* bhh1    = (const float*)d_in[10];
  const float* Wc      = (const float*)d_in[11];
  const float* bc      = (const float*)d_in[12];
  float* out = (float*)d_out;

  char* ws = (char*)d_ws;
  _Float16* w0g = (_Float16*)(ws);                 // 256*256 f16 = 128 KB
  _Float16* w1g = (_Float16*)(ws + 131072);
  _Float16* w2g = (_Float16*)(ws + 262144);
  float*    P0g = (float*)   (ws + 393216);        // 3*256 f32
  float*    b1g = (float*)   (ws + 397312);        // 256 f32

  prep_kernel<<<128, 256, 0, stream>>>(emb, Wih0, Whh0, bih0, bhh0,
                                       Wih1, Whh1, bih1, bhh1,
                                       w0g, w1g, w2g, P0g, b1g);
  rnn_kernel<<<8, 512, 0, stream>>>(tokens, lengths, w0g, w1g, w2g,
                                    P0g, b1g, Wc, bc, out);
}
